// SVDDecomposeTransMatrix_2035814498918
// MI455X (gfx1250) — matrix operand order repaired and checked by reading, not compiled, not run on hardware
//
#include <hip/hip_runtime.h>
#include <hip/hip_bf16.h>

// ---------------------------------------------------------------------------
// MI455X (gfx1250, wave32):  out = ML^T @ (scale o X) @ MR per 64x64 tile.
// Hot loop: v_wmma_f32_16x16x32_bf16, truncation hi/lo bf16 split (3 products
// per k-step ~ fp32 accuracy). X is split ONCE at staging time and stored in
// LDS as packed-bf16 fragments; all WMMA operands load as b128 vectors.
// ---------------------------------------------------------------------------

typedef __attribute__((ext_vector_type(16))) __bf16    v16bf;
typedef __attribute__((ext_vector_type(8)))  float     v8f;
typedef __attribute__((ext_vector_type(8)))  unsigned  v8u;

__device__ __forceinline__ unsigned fbits(float f) { return __builtin_bit_cast(unsigned, f); }
__device__ __forceinline__ float    ubits(unsigned u) { return __builtin_bit_cast(float, u); }

// Truncation split of two f32 (f0 -> low half, f1 -> high half) into
// packed-bf16 hi (.x) / lo (.y) dwords. hi = chop(x); lo = chop(x - hi).
__device__ __forceinline__ uint2 split2(float f0, float f1) {
  unsigned u0 = fbits(f0), u1 = fbits(f1);
  unsigned h0 = u0 & 0xFFFF0000u, h1 = u1 & 0xFFFF0000u;
  unsigned hp = (u0 >> 16) | h1;
  float r0 = f0 - ubits(h0);
  float r1 = f1 - ubits(h1);
  unsigned lp = (fbits(r0) >> 16) | (fbits(r1) & 0xFFFF0000u);
  return make_uint2(hp, lp);
}

// RNE f32 -> bf16 bits (setup kernels only).
__device__ __forceinline__ unsigned short f2bf_rn(float f) {
  unsigned u = fbits(f);
  u += 0x7FFFu + ((u >> 16) & 1u);
  return (unsigned short)(u >> 16);
}
__device__ __forceinline__ float bf2f(unsigned short h) { return ubits((unsigned)h << 16); }

// Load a 16-bf16 fragment stored contiguously per lane (2 x b128).
__device__ __forceinline__ v16bf load_frag_u32(const unsigned* p) {
  const uint4* q = (const uint4*)p;
  uint4 a = q[0], b = q[1];
  v8u v = {a.x, a.y, a.z, a.w, b.x, b.y, b.z, b.w};
  return __builtin_bit_cast(v16bf, v);
}
__device__ __forceinline__ v16bf load_frag_bf16(const unsigned short* p) {
  return load_frag_u32((const unsigned*)p);
}

// 16 contiguous f32 (16B aligned) -> hi/lo bf16 fragments (4 x b128 loads).
__device__ __forceinline__ void frag_from_f32(const float* p, v16bf& H, v16bf& L) {
  const float4* q4 = (const float4*)p;
  float4 a = q4[0], b = q4[1], c = q4[2], d = q4[3];
  v8u h, l;
  uint2 r;
  r = split2(a.x, a.y); h[0] = r.x; l[0] = r.y;
  r = split2(a.z, a.w); h[1] = r.x; l[1] = r.y;
  r = split2(b.x, b.y); h[2] = r.x; l[2] = r.y;
  r = split2(b.z, b.w); h[3] = r.x; l[3] = r.y;
  r = split2(c.x, c.y); h[4] = r.x; l[4] = r.y;
  r = split2(c.z, c.w); h[5] = r.x; l[5] = r.y;
  r = split2(d.x, d.y); h[6] = r.x; l[6] = r.y;
  r = split2(d.z, d.w); h[7] = r.x; l[7] = r.y;
  H = __builtin_bit_cast(v16bf, h);
  L = __builtin_bit_cast(v16bf, l);
}

#define T_STRIDE 20            // f32 lane stride for T fragments (16B aligned)
#define T_FRAG   (32 * T_STRIDE)
#define XB_STRIDE 12           // dword lane stride for packed-bf16 X fragments
#define XB_FRAG  (32 * XB_STRIDE)

// ---------------------------------------------------------------------------
// Kernel 1: Cayley transform of four 64x64 raws; block per matrix, 64 threads.
// Gauss-Jordan solve of (I - X/2) Q = (I + X/2) in LDS (near-identity).
// ---------------------------------------------------------------------------
__global__ __launch_bounds__(64)
void cayley_kernel(const float* __restrict__ ru_l, const float* __restrict__ rv_l,
                   const float* __restrict__ ru_r, const float* __restrict__ rv_r,
                   float* __restrict__ q_out) {
  __shared__ float Ms[64][64];
  __shared__ float Ns[64][64];
  __shared__ float fcol[64];
  const float* raw = (blockIdx.x == 0) ? ru_l
                   : (blockIdx.x == 1) ? rv_l
                   : (blockIdx.x == 2) ? ru_r : rv_r;
  const int t = threadIdx.x;

  for (int i = 0; i < 64; ++i) {
    float xa = (t > i) ? raw[i * 64 + t] : 0.0f;
    float xb = (i > t) ? raw[t * 64 + i] : 0.0f;
    float X  = xa - xb;
    float Id = (i == t) ? 1.0f : 0.0f;
    Ms[i][t] = Id - 0.5f * X;
    Ns[i][t] = Id + 0.5f * X;
  }
  __syncthreads();

  for (int k = 0; k < 64; ++k) {
    float piv = Ms[k][k];
    float mkr = Ms[k][t];
    float nkr = Ns[k][t];
    float fr  = (t == k) ? 0.0f : Ms[t][k];
    __syncthreads();
    float inv = 1.0f / piv;
    float mk = mkr * inv, nk = nkr * inv;
    Ms[k][t] = mk;
    Ns[k][t] = nk;
    fcol[t] = fr;
    __syncthreads();
    for (int i = 0; i < 64; ++i) {
      float f = fcol[i];
      Ms[i][t] -= f * mk;
      Ns[i][t] -= f * nk;
    }
    __syncthreads();
  }
  float* qo = q_out + (size_t)blockIdx.x * 4096;
  for (int i = 0; i < 64; ++i) qo[i * 64 + t] = Ns[i][t];
}

// ---------------------------------------------------------------------------
// Kernel 2: ML = (Ul*dl) Vl^T, MR = (Ur*dr) Vr^T; written PRE-SWIZZLED in
// WMMA fragment order, hi/lo bf16:
//   A-operand (ML^T): frag f = (m>>4)*2 + (k>>5); lane = ((k>>4)&1)*16 + (m&15)
//                     j = k&15;  idx = (f*32 + lane)*16 + j
//   B-operand (MR)  : frag f = (n>>4)*2 + (k>>5); lane = ((k>>4)&1)*16 + (n&15)
//                     j = k&15;                   idx = (f*32 + lane)*16 + j
// ---------------------------------------------------------------------------
__global__ __launch_bounds__(256)
void factors_kernel(const float* __restrict__ q,
                    const float* __restrict__ diag_left,
                    const float* __restrict__ diag_right,
                    const int* __restrict__ inv_t,
                    unsigned short* __restrict__ MLt_hi,
                    unsigned short* __restrict__ MLt_lo,
                    unsigned short* __restrict__ MR_hi,
                    unsigned short* __restrict__ MR_lo) {
  const bool right = (blockIdx.x == 1);
  const float* U = q + (right ? 8192 : 0);
  const float* V = U + 4096;
  const float* d = right ? diag_right : diag_left;
  const bool invf = (*inv_t != 0);

  __shared__ float Us[4096];
  __shared__ float Vs[4096];
  __shared__ float ds[64];
  for (int e = threadIdx.x; e < 4096; e += 256) { Us[e] = U[e]; Vs[e] = V[e]; }
  if (threadIdx.x < 64) {
    float dv = d[threadIdx.x];
    ds[threadIdx.x] = invf ? (1.0f / dv) : dv;
  }
  __syncthreads();

  for (int e = threadIdx.x; e < 4096; e += 256) {
    int i = e >> 6, j = e & 63;                 // M[i][j]
    float s = 0.0f;
    for (int k = 0; k < 64; ++k) s += Us[i * 64 + k] * ds[k] * Vs[j * 64 + k];
    unsigned short hb = f2bf_rn(s);
    unsigned short lb = f2bf_rn(s - bf2f(hb));
    if (right) {
      // MR element (k=i, n=j) -> B-swizzle
      int f  = (j >> 4) * 2 + (i >> 5);
      int ln = ((i >> 4) & 1) * 16 + (j & 15);
      int jj = i & 15;
      int idx = (f * 32 + ln) * 16 + jj;
      MR_hi[idx] = hb;  MR_lo[idx] = lb;
    } else {
      // MLt element (m=j, k=i) -> A-swizzle
      int f  = (j >> 4) * 2 + (i >> 5);
      int ln = ((i >> 4) & 1) * 16 + (j & 15);
      int jj = i & 15;
      int idx = (f * 32 + ln) * 16 + jj;
      MLt_hi[idx] = hb;  MLt_lo[idx] = lb;
    }
  }
}

// ---------------------------------------------------------------------------
// Kernel 3: hot loop. 1 block = 1 tile, 128 threads = 4 waves; wave w owns
// output strip rows [16w,16w+16). X is split hi/lo ONCE at staging and stored
// as packed-bf16 B-fragments in LDS (split not recomputed per wave); GEMM1 B
// operands are pure 2xb128 LDS loads. T strip round-trips through f32 LDS.
// 24 + 24 wmma per wave: AhBh + AlBh + AhBl per k-chunk.
// ---------------------------------------------------------------------------
__global__ __launch_bounds__(128)
void kron_kernel(const float* __restrict__ inp,
                 const float* __restrict__ dscale,
                 const int* __restrict__ inv_t,
                 const unsigned short* __restrict__ MLt_hi,
                 const unsigned short* __restrict__ MLt_lo,
                 const unsigned short* __restrict__ MR_hi,
                 const unsigned short* __restrict__ MR_lo,
                 float* __restrict__ out) {
  __shared__ unsigned Xh[8 * XB_FRAG];      // X hi fragments (packed bf16)
  __shared__ unsigned Xl[8 * XB_FRAG];      // X lo fragments (packed bf16)
  __shared__ float    Ts[4][2 * T_FRAG];    // per-wave T strip, A-frag order

  const int tid  = threadIdx.x;
  const int lane = tid & 31;
  const int wave = tid >> 5;                // strip index 0..3
  const size_t row = blockIdx.x;
  const bool invf = (*inv_t != 0);

  // ---- Stage: load row PAIRS (k=2k2, 2k2+1), scale, split, store packed
  //      hi/lo dwords into B-swizzled LDS (4 b32 scatters per buffer).
  {
    const float4* x4 = (const float4*)(inp + row * 4096);
    const float4* s4 = (const float4*)dscale;
#pragma unroll
    for (int it = 0; it < 4; ++it) {
      int p  = tid + it * 128;              // 0..511 row-pair x col4
      int k2 = p >> 4;                      // row pair 0..31
      int c4 = p & 15;                      // float4 column 0..15
      float4 xa = x4[k2 * 32 + c4];         // row 2*k2   (16 float4 per row)
      float4 xb = x4[k2 * 32 + 16 + c4];    // row 2*k2+1
      float4 sa = s4[k2 * 32 + c4];
      float4 sb = s4[k2 * 32 + 16 + c4];
      float4 ra, rb;
      if (invf) {
        ra.x = xa.x / sa.x; ra.y = xa.y / sa.y; ra.z = xa.z / sa.z; ra.w = xa.w / sa.w;
        rb.x = xb.x / sb.x; rb.y = xb.y / sb.y; rb.z = xb.z / sb.z; rb.w = xb.w / sb.w;
      } else {
        ra.x = xa.x * sa.x; ra.y = xa.y * sa.y; ra.z = xa.z * sa.z; ra.w = xa.w * sa.w;
        rb.x = xb.x * sb.x; rb.y = xb.y * sb.y; rb.z = xb.z * sb.z; rb.w = xb.w * sb.w;
      }
      int k  = k2 * 2;
      int n0 = c4 * 4;
      int f  = (n0 >> 4) * 2 + (k >> 5);
      int ln = ((k >> 4) & 1) * 16 + (n0 & 15);
      int j2 = (k & 15) >> 1;               // packed dword index = k2 & 7
      unsigned* ph = Xh + f * XB_FRAG + ln * XB_STRIDE + j2;
      unsigned* pl = Xl + f * XB_FRAG + ln * XB_STRIDE + j2;
      uint2 r0 = split2(ra.x, rb.x); ph[0]             = r0.x; pl[0]             = r0.y;
      uint2 r1 = split2(ra.y, rb.y); ph[XB_STRIDE]     = r1.x; pl[XB_STRIDE]     = r1.y;
      uint2 r2 = split2(ra.z, rb.z); ph[2 * XB_STRIDE] = r2.x; pl[2 * XB_STRIDE] = r2.y;
      uint2 r3 = split2(ra.w, rb.w); ph[3 * XB_STRIDE] = r3.x; pl[3 * XB_STRIDE] = r3.y;
    }
  }
  __syncthreads();

  // C-layout coords: vgpr g, lane -> M = 8*(lane>>4)+g, N = lane&15
  const int cm = (lane >> 4) * 8;
  const int cn = lane & 15;

  // ---- A fragments of GEMM1: MLt strip (pre-swizzled bf16, 2 b128 each) ----
  v16bf A1h[2], A1l[2];
#pragma unroll
  for (int kc = 0; kc < 2; ++kc) {
    int base = ((wave * 2 + kc) * 32 + lane) * 16;
    A1h[kc] = load_frag_bf16(MLt_hi + base);
    A1l[kc] = load_frag_bf16(MLt_lo + base);
  }

  // ---- GEMM1: T strip = MLt[strip] @ X (B frags: pure b128 LDS loads) ----
  v8f acc[4];
#pragma unroll
  for (int nt = 0; nt < 4; ++nt) {
    v8f c = {};
#pragma unroll
    for (int kc = 0; kc < 2; ++kc) {
      int fb = (nt * 2 + kc) * XB_FRAG + lane * XB_STRIDE;
      v16bf Bh = load_frag_u32(Xh + fb);
      v16bf Bl = load_frag_u32(Xl + fb);
      c = __builtin_amdgcn_wmma_f32_16x16x32_bf16(false, A1h[kc], false, Bh, (short)0, c, false, false);
      c = __builtin_amdgcn_wmma_f32_16x16x32_bf16(false, A1l[kc], false, Bh, (short)0, c, false, false);
      c = __builtin_amdgcn_wmma_f32_16x16x32_bf16(false, A1h[kc], false, Bl, (short)0, c, false, false);
    }
    acc[nt] = c;
  }

  // ---- Scatter C-layout accumulators into A-swizzled wave-private LDS ----
  // T element (m = cm+g, k = nt*16+cn):
  //   kc2 = nt>>1; j = cn; l = (nt&1)*16 + m
#pragma unroll
  for (int nt = 0; nt < 4; ++nt) {
    float* base = Ts[wave] + (nt >> 1) * T_FRAG + cn
                + ((nt & 1) * 16 + cm) * T_STRIDE;
#pragma unroll
    for (int g = 0; g < 8; ++g)
      base[g * T_STRIDE] = acc[nt][g];
  }
  // same-wave LDS RAW: DS ops are in order / compiler inserts s_wait_dscnt

  // ---- A fragments of GEMM2 from T strip (4 b128 LDS reads each) ----
  v16bf A2h[2], A2l[2];
#pragma unroll
  for (int kc = 0; kc < 2; ++kc)
    frag_from_f32(Ts[wave] + kc * T_FRAG + lane * T_STRIDE, A2h[kc], A2l[kc]);

  // ---- GEMM2: out strip = T_strip @ MR (pre-swizzled bf16 B operand) ----
  float* orow = out + row * 4096;
#pragma unroll
  for (int nt = 0; nt < 4; ++nt) {
    v8f c = {};
#pragma unroll
    for (int kc = 0; kc < 2; ++kc) {
      int base = ((nt * 2 + kc) * 32 + lane) * 16;
      v16bf Bh = load_frag_bf16(MR_hi + base);
      v16bf Bl = load_frag_bf16(MR_lo + base);
      c = __builtin_amdgcn_wmma_f32_16x16x32_bf16(false, A2h[kc], false, Bh, (short)0, c, false, false);
      c = __builtin_amdgcn_wmma_f32_16x16x32_bf16(false, A2l[kc], false, Bh, (short)0, c, false, false);
      c = __builtin_amdgcn_wmma_f32_16x16x32_bf16(false, A2h[kc], false, Bl, (short)0, c, false, false);
    }
#pragma unroll
    for (int g = 0; g < 8; ++g)
      orow[(wave * 16 + cm + g) * 64 + nt * 16 + cn] = c[g];
  }
}

// ---------------------------------------------------------------------------
extern "C" void kernel_launch(void* const* d_in, const int* in_sizes, int n_in,
                              void* d_out, int out_size, void* d_ws, size_t ws_size,
                              hipStream_t stream) {
  (void)n_in; (void)out_size; (void)ws_size;
  const float* inp  = (const float*)d_in[0];
  const float* ru_l = (const float*)d_in[1];
  const float* rv_l = (const float*)d_in[2];
  const float* dl   = (const float*)d_in[3];
  const float* ru_r = (const float*)d_in[4];
  const float* rv_r = (const float*)d_in[5];
  const float* dr   = (const float*)d_in[6];
  const float* dsc  = (const float*)d_in[7];
  const int*   invt = (const int*)d_in[8];
  float* out = (float*)d_out;

  // Workspace: q[4][4096] f32, then 4 x 4096 bf16 swizzled factor buffers.
  float* q = (float*)d_ws;
  unsigned short* MLt_hi = (unsigned short*)(q + 16384);
  unsigned short* MLt_lo = MLt_hi + 4096;
  unsigned short* MR_hi  = MLt_lo + 4096;
  unsigned short* MR_lo  = MR_hi + 4096;

  cayley_kernel<<<4, 64, 0, stream>>>(ru_l, rv_l, ru_r, rv_r, q);
  factors_kernel<<<2, 256, 0, stream>>>(q, dl, dr, invt,
                                        MLt_hi, MLt_lo, MR_hi, MR_lo);
  const int nrows = in_sizes[0] / 4096;     // 8192
  kron_kernel<<<nrows, 128, 0, stream>>>(inp, dsc, invt,
                                         MLt_hi, MLt_lo, MR_hi, MR_lo, out);
}
